// SADM_A_21534966022193
// MI455X (gfx1250) — compile-verified, matrix operand order checked by reading, not run on hardware
//
#include <hip/hip_runtime.h>
#include <hip/hip_bf16.h>
#include <stdint.h>

// B=32, C_IN=C_OUT=256, H=W=32, HW=1024
// ws layout (bytes):
//   xT  : [32][1024][256] bf16  @ 0          (16,777,216)
//   wbf : [256][256]      bf16  @ 16777216   (131,072)
//   vT  : [32][256][1024] bf16  @ 16908288   (16,777,216)

typedef __attribute__((ext_vector_type(16))) __bf16 v16bf;
typedef __attribute__((ext_vector_type(8)))  float  v8f;
typedef __attribute__((ext_vector_type(4)))  int    v4i;

__device__ __forceinline__ unsigned short f2bf(float f) {
  union { float f; uint32_t u; } c; c.f = f;
  uint32_t u = c.u;
  u += 0x7fffu + ((u >> 16) & 1u);   // round-to-nearest-even
  return (unsigned short)(u >> 16);
}

// ---- async global -> LDS copy (16B granules), with guaranteed-compile fallback
#if __has_builtin(__builtin_amdgcn_global_load_async_to_lds_b128)
#define ASYNC_LDS 1
typedef __attribute__((address_space(1))) v4i* gv4p;
typedef __attribute__((address_space(3))) v4i* lv4p;
__device__ __forceinline__ void copy16(const void* g, void* l) {
  __builtin_amdgcn_global_load_async_to_lds_b128((gv4p)(void*)g, (lv4p)l, 0, 0);
}
#define WAIT_ASYNC_8() __builtin_amdgcn_s_wait_asynccnt(8)
#define WAIT_ASYNC_0() __builtin_amdgcn_s_wait_asynccnt(0)
#else
#define ASYNC_LDS 0
__device__ __forceinline__ void copy16(const void* g, void* l) {
  *(uint4*)l = *(const uint4*)g;     // sync fallback: global->reg->LDS
}
#define WAIT_ASYNC_8() ((void)0)
#define WAIT_ASYNC_0() ((void)0)
#endif

// ---- Kernel 1: x (b,c,pos) f32 -> xT (b,pos,c) bf16, tiled transpose ----
__global__ void k_xT(const float* __restrict__ x, unsigned short* __restrict__ xT) {
  __shared__ float t[16][17];
  int bid = blockIdx.x;
  int b = bid >> 10, rem = bid & 1023, ct = rem >> 6, pt = rem & 63;
  int tx = threadIdx.x & 15, ty = threadIdx.x >> 4;
  t[ty][tx] = x[(((size_t)b * 256 + ct * 16 + ty) << 10) + pt * 16 + tx];
  __syncthreads();
  xT[(((size_t)b << 10) + pt * 16 + ty) * 256 + ct * 16 + tx] = f2bf(t[tx][ty]);
}

// ---- Kernel 2: conv_w f32 -> bf16 ----
__global__ void k_w(const float* __restrict__ w, unsigned short* __restrict__ wbf) {
  int i = blockIdx.x * 256 + threadIdx.x;
  wbf[i] = f2bf(w[i]);
}

// ---- Kernel 3: vT(oc,pos) = W(oc,c) @ x(c,pos) + bias, bf16 out, WMMA bf16 ----
__global__ void k_vT(const unsigned short* __restrict__ xT,
                     const unsigned short* __restrict__ wbf,
                     const float* __restrict__ bias,
                     unsigned short* __restrict__ vT) {
  int lane = threadIdx.x & 31;
  int wid  = blockIdx.x * 8 + (threadIdx.x >> 5);
  int b = wid >> 8, rem = wid & 255, mt = rem >> 4, ng = rem & 15;
  int ln = lane & 15, lh = lane >> 4;
  const unsigned short* xTb = xT + ((size_t)b << 18);

  v16bf a[8];
#pragma unroll
  for (int kk = 0; kk < 8; ++kk)
    a[kk] = *(const v16bf*)(wbf + (mt * 16 + ln) * 256 + kk * 32 + lh * 16);
  float bv[8];
#pragma unroll
  for (int r = 0; r < 8; ++r) bv[r] = bias[mt * 16 + r + lh * 8];

  unsigned short* dst = vT + ((size_t)b << 18);
#pragma unroll
  for (int s = 0; s < 4; ++s) {
    int nb = ng * 64 + s * 16;
    v8f c = {};
#pragma unroll
    for (int kk = 0; kk < 8; ++kk) {
      v16bf bf = *(const v16bf*)(xTb + (nb + ln) * 256 + kk * 32 + lh * 16);
      c = __builtin_amdgcn_wmma_f32_16x16x32_bf16(false, a[kk], false, bf,
                                                  (short)0, c, false, false);
    }
#pragma unroll
    for (int r = 0; r < 8; ++r) {
      int oc = mt * 16 + r + lh * 8;
      dst[((size_t)oc << 10) + nb + ln] = f2bf(c[r] + bv[r]);
    }
  }
}

// ---- Kernel 4: flash-attention + scrambled reshape + 2x2 maxpool ----
// LDS map (bytes): kbuf[2] @0 (2x16K), vbuf[2] @32768 (2x16K),
//                  pbuf @65536 + wave*1024, obuf @73728 + wave*16384. Total 204800.
__device__ __forceinline__ void stage_tiles(const unsigned short* xTb,
                                            const unsigned short* vTb,
                                            char* smem, int buf, int jt, int tid) {
  char* kdst = smem + buf * 16384;
  char* vdst = smem + 32768 + buf * 16384;
  const char* ksrc = (const char*)(xTb + (size_t)jt * 32 * 256); // contiguous 16KB slab
#pragma unroll
  for (int i = 0; i < 4; ++i) {
    int q = i * 256 + tid;
    copy16(ksrc + q * 16, kdst + q * 16);
  }
#pragma unroll
  for (int i = 0; i < 4; ++i) {
    int q = i * 256 + tid;
    int oc = q >> 2, part = q & 3;            // 4 x 16B per oc-row of 64B
    const char* src = (const char*)(vTb + ((size_t)oc << 10) + jt * 32) + part * 16;
    copy16(src, vdst + oc * 64 + part * 16);
  }
}

__global__ void k_attn(const unsigned short* __restrict__ xT,
                       const unsigned short* __restrict__ vT,
                       float* __restrict__ out) {
  extern __shared__ char smem[];
  int tid  = threadIdx.x;
  int lane = tid & 31;
  int wave = tid >> 5;
  int b = blockIdx.x >> 3;
  int Mbase = ((blockIdx.x & 7) * 8 + wave) * 16;
  int ln = lane & 15, lh = lane >> 4;
  const unsigned short* xTb = xT + ((size_t)b << 18);
  const unsigned short* vTb = vT + ((size_t)b << 18);
  unsigned short* pbuf = (unsigned short*)(smem + 65536 + wave * 1024);
  float*          obuf = (float*)(smem + 73728 + wave * 16384);

  // Resident A fragments: Q rows (gram symmetry: same buffer serves K)
  v16bf a[8];
#pragma unroll
  for (int kk = 0; kk < 8; ++kk)
    a[kk] = *(const v16bf*)(xTb + (Mbase + ln) * 256 + kk * 32 + lh * 16);

  v8f acc[16];
#pragma unroll
  for (int t = 0; t < 16; ++t) acc[t] = (v8f){};
  float m[8], l[8];
#pragma unroll
  for (int r = 0; r < 8; ++r) { m[r] = -3.0e38f; l[r] = 0.0f; }

  stage_tiles(xTb, vTb, smem, 0, 0, tid);     // prefetch jt=0

  for (int jt = 0; jt < 32; ++jt) {
    int cur = jt & 1;
    if (jt < 31) {
      stage_tiles(xTb, vTb, smem, cur ^ 1, jt + 1, tid);  // overlap DMA with compute
      WAIT_ASYNC_8();                                      // current tile complete
    } else {
      WAIT_ASYNC_0();
    }
    __syncthreads();                                       // tile visible to all waves

    const unsigned short* kb = (const unsigned short*)(smem + cur * 16384);
    const unsigned short* vb = (const unsigned short*)(smem + 32768 + cur * 16384);

    v8f S[2];
#pragma unroll
    for (int s = 0; s < 2; ++s) {
      v8f c = {};
#pragma unroll
      for (int kk = 0; kk < 8; ++kk) {
        v16bf bf = *(const v16bf*)(kb + (s * 16 + ln) * 256 + kk * 32 + lh * 16);
        c = __builtin_amdgcn_wmma_f32_16x16x32_bf16(false, a[kk], false, bf,
                                                    (short)0, c, false, false);
      }
      S[s] = c;
    }
    // online softmax (16 lanes of each half hold one row's 16 cols)
#pragma unroll
    for (int r = 0; r < 8; ++r) {
      float tmax = fmaxf(S[0][r], S[1][r]);
#pragma unroll
      for (int off = 1; off < 16; off <<= 1)
        tmax = fmaxf(tmax, __shfl_xor(tmax, off, 32));
      float mn = fmaxf(m[r], tmax);
      float sc = __expf(m[r] - mn);
      float p0 = __expf(S[0][r] - mn);
      float p1 = __expf(S[1][r] - mn);
      S[0][r] = p0; S[1][r] = p1;
      float rs = p0 + p1;
#pragma unroll
      for (int off = 1; off < 16; off <<= 1)
        rs += __shfl_xor(rs, off, 32);
      l[r] = l[r] * sc + rs;
      m[r] = mn;
#pragma unroll
      for (int t = 0; t < 16; ++t) acc[t][r] *= sc;
    }
    // stage P (16x32) to LDS in A-fragment order
#pragma unroll
    for (int r = 0; r < 8; ++r) {
      int row = r + lh * 8;
      pbuf[row * 32 + ln]      = f2bf(S[0][r]);
      pbuf[row * 32 + 16 + ln] = f2bf(S[1][r]);
    }
    v16bf pa = *(const v16bf*)(pbuf + ln * 32 + lh * 16);
    // acc += P @ V (16 N-tiles covering 256 out channels), V frags from LDS
#pragma unroll
    for (int t = 0; t < 16; ++t) {
      v16bf bf = *(const v16bf*)(vb + (t * 16 + ln) * 32 + lh * 16);
      acc[t] = __builtin_amdgcn_wmma_f32_16x16x32_bf16(false, pa, false, bf,
                                                       (short)0, acc[t], false, false);
    }
    __syncthreads();    // all waves done reading cur before it is re-staged
  }

  // epilogue: normalize, stage 16x256 row block to LDS
  float inv[8];
#pragma unroll
  for (int r = 0; r < 8; ++r) inv[r] = 1.0f / l[r];
#pragma unroll
  for (int t = 0; t < 16; ++t)
#pragma unroll
    for (int r = 0; r < 8; ++r)
      obuf[(r + lh * 8) * 256 + t * 16 + ln] = acc[t][r] * inv[r];

  // scrambled reshape + 2x2 maxpool: row i feeds (c'=i>>2, ph=4*(i&3)+p, pw);
  // the 4 pooled sources are cols {(2p+dh)*32 + 2pw+dw} of the SAME row.
  float* outb = out + ((size_t)b << 16);
#pragma unroll
  for (int u = 0; u < 32; ++u) {
    int idx = u * 32 + lane;          // [row(4b) | p(2b) | pw(4b)]
    int pw = idx & 15, p = (idx >> 4) & 3, row = idx >> 6;
    int ob = (2 * p) * 32 + 2 * pw;
    float v0 = obuf[row * 256 + ob];
    float v1 = obuf[row * 256 + ob + 1];
    float v2 = obuf[row * 256 + ob + 32];
    float v3 = obuf[row * 256 + ob + 33];
    float mx = fmaxf(fmaxf(v0, v1), fmaxf(v2, v3));
    int i  = Mbase + row;
    int cp = i >> 2;
    int ph = ((i & 3) << 2) + p;
    outb[(cp << 8) + (ph << 4) + pw] = mx;
  }
}

extern "C" void kernel_launch(void* const* d_in, const int* in_sizes, int n_in,
                              void* d_out, int out_size, void* d_ws, size_t ws_size,
                              hipStream_t stream) {
  const float* x    = (const float*)d_in[0];
  const float* w    = (const float*)d_in[1];
  const float* bias = (const float*)d_in[2];
  float* out = (float*)d_out;

  unsigned short* xT  = (unsigned short*)d_ws;
  unsigned short* wbf = (unsigned short*)((char*)d_ws + 16777216);
  unsigned short* vT  = (unsigned short*)((char*)d_ws + 16777216 + 131072);

  k_xT  <<<32768, 256, 0, stream>>>(x, xT);
  k_w   <<<256,   256, 0, stream>>>(w, wbf);
  k_vT  <<<1024,  256, 0, stream>>>(xT, wbf, bias, vT);
  k_attn<<<256,   256, 204800, stream>>>(xT, vT, out);
}
